// DeepSupportConvex_17592186045118
// MI455X (gfx1250) — matrix operand order repaired and checked by reading, not compiled, not run on hardware
//
#include <hip/hip_runtime.h>
#include <hip/hip_bf16.h>

// ---------------------------------------------------------------------------
// DeepSupportConvex on MI455X (gfx1250):
//   pack_weights : |W_hid| (+transpose) -> bf16, pre-swizzled into WMMA
//                  B-fragment lane order (one contiguous 32B load per lane).
//   icnn_grad    : fused ICNN forward + backward per 16-point tile, all four
//                  512x512 contractions on v_wmma_f32_16x16x32_bf16.
//   finalize     : dots vs. original direction + stable top-4 sort + gather.
// ---------------------------------------------------------------------------

typedef __attribute__((ext_vector_type(16))) __bf16 v16bf;
typedef __attribute__((ext_vector_type(8)))  __bf16 v8bf;
typedef __attribute__((ext_vector_type(8)))  float  v8f;

#define WIDTH 512
#define NEG_SLOPE 0.5f

// ---- fragment loaders ------------------------------------------------------

// A fragment (16x32 bf16, M x K) from LDS row-major [16][512].
// ISA layout: lanes 0-15 -> K 0..7 (VGPR0-3), 16..23 (VGPR4-7);
//             lanes 16-31 -> K 8..15, 24..31.  Row m = lane % 16.
__device__ __forceinline__ v16bf load_a_frag(const __bf16* Abuf, int kt, int lane) {
    int m    = lane & 15;
    int kgrp = (lane < 16) ? 0 : 8;
    const __bf16* p = Abuf + m * WIDTH + kt * 32 + kgrp;
    v8bf lo = *(const v8bf*)(p);        // k + 0..7   -> VGPRs 0-3
    v8bf hi = *(const v8bf*)(p + 16);   // k + 16..23 -> VGPRs 4-7
    return __builtin_shufflevector(lo, hi, 0,1,2,3,4,5,6,7,8,9,10,11,12,13,14,15);
}

// B fragment (32x16 bf16, K x N) from pre-swizzled global weights:
// fragment id = kt*32 + nt, each lane's 16 bf16 stored contiguously.
__device__ __forceinline__ v16bf load_b_frag(const __bf16* Bfrag, int kt, int nt, int lane) {
    const __bf16* p = Bfrag + (((size_t)(kt * 32 + nt)) << 9) + lane * 16;
    return *(const v16bf*)p;
}

// ---- fused GEMM tile (one wave: 16 rows x 64 cols, K = 512) ----------------
// MODE 0: forward hidden layer: z = A@B + d.W_in[l]; write h=leaky(z), mask.
// MODE 1: last forward layer:   z = A@B + d.W_in[2]; write g2 = |w_out|*s'(z).
// MODE 2: backward layer:       t = A@B;             write g  = t * maskIn.
template<int MODE>
__device__ __forceinline__ void gemm_tile(const __bf16* __restrict__ Abuf,
                                          const __bf16* __restrict__ Bfrag,
                                          const float*  __restrict__ dirs,   // LDS [16][4]
                                          const float*  __restrict__ Win_l,  // global, 3x512
                                          const float*  __restrict__ wout,   // global, 512
                                          const __bf16* __restrict__ maskIn, // LDS [16][512]
                                          __bf16* __restrict__ Hout,
                                          __bf16* __restrict__ Mout,
                                          int lane, int wave) {
    v8f acc[4] = {};
#pragma unroll 4
    for (int kt = 0; kt < 16; ++kt) {
        v16bf a = load_a_frag(Abuf, kt, lane);
#pragma unroll
        for (int t = 0; t < 4; ++t) {
            v16bf b = load_b_frag(Bfrag, kt, wave * 4 + t, lane);
            acc[t] = __builtin_amdgcn_wmma_f32_16x16x32_bf16(
                false, a, false, b, (short)0, acc[t], false, false);
        }
    }
    int mofs = (lane >= 16) ? 8 : 0;
#pragma unroll
    for (int t = 0; t < 4; ++t) {
        int n = wave * 64 + t * 16 + (lane & 15);
        float w0 = 0.f, w1 = 0.f, w2 = 0.f, wo = 0.f;
        if (MODE == 0 || MODE == 1) {
            w0 = Win_l[n]; w1 = Win_l[WIDTH + n]; w2 = Win_l[2 * WIDTH + n];
        }
        if (MODE == 1) wo = fabsf(wout[n]);
#pragma unroll
        for (int r = 0; r < 8; ++r) {
            int m = r + mofs;
            float z = acc[t][r];
            if (MODE == 0 || MODE == 1)
                z += dirs[m * 4 + 0] * w0 + dirs[m * 4 + 1] * w1 + dirs[m * 4 + 2] * w2;
            if (MODE == 0) {
                Hout[m * WIDTH + n] = (__bf16)(z >= 0.f ? z : NEG_SLOPE * z);
                Mout[m * WIDTH + n] = (__bf16)(z >= 0.f ? 1.0f : NEG_SLOPE);
            } else if (MODE == 1) {
                Hout[m * WIDTH + n] = (__bf16)(wo * (z >= 0.f ? 1.0f : NEG_SLOPE));
            } else {
                float mk = (float)maskIn[m * WIDTH + n];
                Hout[m * WIDTH + n] = (__bf16)(z * mk);
            }
        }
    }
}

// ---- grad accumulation: grad[r][x] += sum_c W_in[l][x][c] * g[r][c] --------
__device__ __forceinline__ void grad_accum(const __bf16* __restrict__ G,
                                           const float*  __restrict__ Win_l,
                                           float* red,    // LDS 16*16*3
                                           float* grad,   // LDS 16*3
                                           int tid) {
    int row = tid >> 4;            // 0..15
    int grp = tid & 15;            // 0..15  -> columns [grp*32, grp*32+32)
    float s0 = 0.f, s1 = 0.f, s2 = 0.f;
    int c0 = grp * 32;
#pragma unroll 8
    for (int c = c0; c < c0 + 32; ++c) {
        float g = (float)G[row * WIDTH + c];
        s0 += g * Win_l[c];
        s1 += g * Win_l[WIDTH + c];
        s2 += g * Win_l[2 * WIDTH + c];
    }
    red[(row * 16 + grp) * 3 + 0] = s0;
    red[(row * 16 + grp) * 3 + 1] = s1;
    red[(row * 16 + grp) * 3 + 2] = s2;
    __syncthreads();
    if (tid < 48) {
        int r = tid / 3, x = tid % 3;
        float s = 0.f;
#pragma unroll
        for (int g = 0; g < 16; ++g) s += red[(r * 16 + g) * 3 + x];
        grad[r * 3 + x] += s;
    }
    __syncthreads();
}

// ---- kernel 1: weight packing ---------------------------------------------
__global__ void pack_weights_kernel(const float* __restrict__ W_hid,
                                    __bf16* __restrict__ Wh0,  __bf16* __restrict__ Wh1,
                                    __bf16* __restrict__ Wh0T, __bf16* __restrict__ Wh1T) {
    int idx = blockIdx.x * blockDim.x + threadIdx.x;   // over 512*512
    if (idx >= WIDTH * WIDTH) return;
    int k = idx >> 9, n = idx & (WIDTH - 1);
    int kt = k >> 5, kr = k & 31;
    int nt = n >> 4, nc = n & 15;
    int lane   = (((kr >> 3) & 1) ? 16 : 0) + nc;
    int klocal = (kr & 7) + ((kr >> 4) << 3);
    size_t dst = (((size_t)(kt * 32 + nt)) << 9) + (size_t)lane * 16 + klocal;
    const size_t L = (size_t)WIDTH * WIDTH;
    Wh0 [dst] = (__bf16)fabsf(W_hid[(size_t)k * WIDTH + n]);
    Wh1 [dst] = (__bf16)fabsf(W_hid[L + (size_t)k * WIDTH + n]);
    Wh0T[dst] = (__bf16)fabsf(W_hid[(size_t)n * WIDTH + k]);       // B = Wh0^T
    Wh1T[dst] = (__bf16)fabsf(W_hid[L + (size_t)n * WIDTH + k]);   // B = Wh1^T
}

// ---- kernel 2: fused ICNN forward + backward (16 points / WG) --------------
__global__ __launch_bounds__(256)
void icnn_grad_kernel(const float* __restrict__ directions,
                      const float* __restrict__ perturbations,
                      const float* __restrict__ W_in,
                      const float* __restrict__ w_out,
                      const __bf16* __restrict__ Wh0,  const __bf16* __restrict__ Wh1,
                      const __bf16* __restrict__ Wh0T, const __bf16* __restrict__ Wh1T,
                      float* __restrict__ vertices) {
    __shared__ __align__(16) __bf16 bufX[16 * WIDTH];  // activations / grads (ping)
    __shared__ __align__(16) __bf16 bufY[16 * WIDTH];  // activations / grads (pong)
    __shared__ __align__(16) __bf16 m0[16 * WIDTH];    // leaky' mask, layer 0
    __shared__ __align__(16) __bf16 m1[16 * WIDTH];    // leaky' mask, layer 1
    __shared__ float dirs[16 * 4];
    __shared__ float red[16 * 16 * 3];
    __shared__ float grad[16 * 3];

    int tid  = threadIdx.x;
    int lane = tid & 31;
    int wave = tid >> 5;
    int row0 = blockIdx.x * 16;

    if (tid < 16) {
        int p = row0 + tid;
        int b = p >> 2, q = p & 3;
        float dx = directions[b * 3 + 0] + perturbations[q * 3 + 0];
        float dy = directions[b * 3 + 1] + perturbations[q * 3 + 1];
        float dz = directions[b * 3 + 2] + perturbations[q * 3 + 2];
        float inv = rsqrtf(dx * dx + dy * dy + dz * dz);
        dirs[tid * 4 + 0] = dx * inv;
        dirs[tid * 4 + 1] = dy * inv;
        dirs[tid * 4 + 2] = dz * inv;
        dirs[tid * 4 + 3] = 0.f;
    }
    if (tid < 48) grad[tid] = 0.f;
    __syncthreads();

    // layer 0: z0 = d @ W_in[0]  (tiny, VALU); h0 -> bufX, mask -> m0
    for (int idx = tid; idx < 16 * WIDTH; idx += 256) {
        int r = idx >> 9, c = idx & (WIDTH - 1);
        float z = dirs[r * 4 + 0] * W_in[c]
                + dirs[r * 4 + 1] * W_in[WIDTH + c]
                + dirs[r * 4 + 2] * W_in[2 * WIDTH + c];
        bufX[idx] = (__bf16)(z >= 0.f ? z : NEG_SLOPE * z);
        m0[idx]   = (__bf16)(z >= 0.f ? 1.0f : NEG_SLOPE);
    }
    __syncthreads();

    // layer 1: h1 = leaky(h0 @ |Wh0| + d @ W_in[1]) -> bufY, mask -> m1
    gemm_tile<0>(bufX, Wh0, dirs, W_in + 1536, nullptr, nullptr, bufY, m1, lane, wave);
    __syncthreads();

    // layer 2: g2 = |w_out| * leaky'(h1 @ |Wh1| + d @ W_in[2]) -> bufX
    gemm_tile<1>(bufY, Wh1, dirs, W_in + 3072, w_out, nullptr, bufX, nullptr, lane, wave);
    __syncthreads();

    grad_accum(bufX, W_in + 3072, red, grad, tid);      // += W_in[2] @ g2

    // g1 = (g2 @ |Wh1|^T) * mask1 -> bufY
    gemm_tile<2>(bufX, Wh1T, dirs, nullptr, nullptr, m1, bufY, nullptr, lane, wave);
    __syncthreads();
    grad_accum(bufY, W_in + 1536, red, grad, tid);      // += W_in[1] @ g1

    // g0 = (g1 @ |Wh0|^T) * mask0 -> bufX
    gemm_tile<2>(bufY, Wh0T, dirs, nullptr, nullptr, m0, bufX, nullptr, lane, wave);
    __syncthreads();
    grad_accum(bufX, W_in, red, grad, tid);             // += W_in[0] @ g0

    if (tid < 48) vertices[(size_t)row0 * 3 + tid] = grad[tid];
}

// ---- kernel 3: dots + stable top-4 sort + gather ---------------------------
__global__ void finalize_topk_kernel(const float* __restrict__ directions,
                                     const float* __restrict__ vertices,
                                     float* __restrict__ out, int B) {
    int b = blockIdx.x * blockDim.x + threadIdx.x;
    if (b >= B) return;
    float dx = directions[b * 3 + 0], dy = directions[b * 3 + 1], dz = directions[b * 3 + 2];
    float vx[4], vy[4], vz[4], dot[4];
#pragma unroll
    for (int q = 0; q < 4; ++q) {
        size_t base = ((size_t)b * 4 + q) * 3;
        vx[q] = vertices[base + 0];
        vy[q] = vertices[base + 1];
        vz[q] = vertices[base + 2];
        dot[q] = vx[q] * dx + vy[q] * dy + vz[q] * dz;
    }
    int ord[4] = {0, 1, 2, 3};
    // stable bubble sort, descending (matches lax.top_k tie-break: lower idx first)
#pragma unroll
    for (int i = 0; i < 3; ++i)
#pragma unroll
        for (int j = 0; j < 3 - i; ++j)
            if (dot[ord[j]] < dot[ord[j + 1]]) { int t = ord[j]; ord[j] = ord[j + 1]; ord[j + 1] = t; }
#pragma unroll
    for (int s = 0; s < 4; ++s) {
        int q = ord[s];
        size_t base = ((size_t)b * 4 + s) * 3;
        out[base + 0] = vx[q];
        out[base + 1] = vy[q];
        out[base + 2] = vz[q];
    }
}

// ---------------------------------------------------------------------------
extern "C" void kernel_launch(void* const* d_in, const int* in_sizes, int n_in,
                              void* d_out, int out_size, void* d_ws, size_t ws_size,
                              hipStream_t stream) {
    const float* directions    = (const float*)d_in[0];  // (B,3)
    const float* perturbations = (const float*)d_in[1];  // (4,3)
    const float* W_in          = (const float*)d_in[2];  // (3,3,512)
    const float* W_hid         = (const float*)d_in[3];  // (2,512,512)
    const float* w_out         = (const float*)d_in[4];  // (512,)

    int B = in_sizes[0] / 3;          // 8192
    int N = B * 4;                    // 32768 points

    char* ws = (char*)d_ws;
    const size_t WBYTES = (size_t)WIDTH * WIDTH * sizeof(__bf16);  // 512 KB each
    __bf16* Wh0   = (__bf16*)(ws + 0 * WBYTES);
    __bf16* Wh1   = (__bf16*)(ws + 1 * WBYTES);
    __bf16* Wh0T  = (__bf16*)(ws + 2 * WBYTES);
    __bf16* Wh1T  = (__bf16*)(ws + 3 * WBYTES);
    float* vertices = (float*)(ws + 4 * WBYTES);                   // N*3 f32

    pack_weights_kernel<<<(WIDTH * WIDTH + 255) / 256, 256, 0, stream>>>(
        W_hid, Wh0, Wh1, Wh0T, Wh1T);

    icnn_grad_kernel<<<N / 16, 256, 0, stream>>>(
        directions, perturbations, W_in, w_out, Wh0, Wh1, Wh0T, Wh1T, vertices);

    finalize_topk_kernel<<<(B + 255) / 256, 256, 0, stream>>>(
        directions, vertices, (float*)d_out, B);
}
